// UniAttention_21492016349296
// MI455X (gfx1250) — matrix operand order repaired and checked by reading, not compiled, not run on hardware
//
#include <hip/hip_runtime.h>

// ---------------------------------------------------------------------------
// Flash-attention (length-masked) for MI455X / gfx1250, bf16 WMMA + fp32 acc.
// B=16, LQ=LK=2048, D=1024, fp32 in/out.
// One workgroup (256 threads = 8 wave32) owns a 32-query tile of one batch:
//   - Q tile staged to LDS as bf16 (32x1024)
//   - O accumulator kept in LDS as f32 (32x1024)
//   - K streamed from global; each wave owns a 16-key strip of a 128-key
//     block and reuses the converted K B-operand for BOTH 16-row M subtiles
//     (halves K traffic + f32->bf16 conversions per WMMA)
//   - V staged transposed to LDS as bf16 in four 256-column quarter passes
//   - softmax: score tile round-trips LDS, row-parallel max/exp/sum
// ---------------------------------------------------------------------------

typedef __attribute__((ext_vector_type(16))) __bf16 v16bf;
typedef __attribute__((ext_vector_type(2)))  __bf16 v2bf;
typedef __attribute__((ext_vector_type(2)))  float  v2f;
typedef __attribute__((ext_vector_type(8)))  float  v8f;

union BF16x16 {
  v16bf v;
  unsigned int w[8];
  uint4 q[2];
};

// pack two floats into two bf16 (low = a, high = b): lowers to v_cvt_pk_bf16_f32
__device__ __forceinline__ unsigned int pk2bf(float a, float b) {
  v2f x = {a, b};
  v2bf r = __builtin_convertvector(x, v2bf);
  return __builtin_bit_cast(unsigned int, r);
}

constexpr int BATCH = 16;
constexpr int LQ    = 2048;
constexpr int LK    = 2048;
constexpr int DIM   = 1024;
constexpr int BM    = 32;     // query rows per workgroup
constexpr int BN    = 128;    // keys per block
constexpr int VQ    = 256;    // O columns per PV quarter pass

// padded LDS strides (elements)
constexpr int QSTR = DIM + 8;    // bf16, 1032 -> 516 dw = 4 mod 64 banks
constexpr int OSTR = DIM + 4;    // f32,  1028 -> 4 mod 64 banks
constexpr int VSTR = BN + 8;     // bf16, 136  -> 68 dw = 4 mod 64 banks
constexpr int PSTR = BN + 8;     // bf16, 136  (row stride 272B, 16B aligned)
constexpr int SSTR = BN + 4;     // f32,  132  -> 4 mod 64 banks

// LDS byte offsets
constexpr int OFF_Q  = 0;
constexpr int SZ_Q   = BM * QSTR * 2;              // 66048
constexpr int OFF_O  = OFF_Q + SZ_Q;
constexpr int SZ_O   = BM * OSTR * 4;              // 131584
constexpr int OFF_V  = OFF_O + SZ_O;
constexpr int SZ_V   = VQ * VSTR * 2;              // 69632
constexpr int OFF_P  = OFF_V + SZ_V;
constexpr int SZ_P   = BM * PSTR * 2;              // 8704
constexpr int OFF_S  = OFF_P + SZ_P;
constexpr int SZ_S   = BM * SSTR * 4;              // 16896
constexpr int OFF_ST = OFF_S + SZ_S;
constexpr int SZ_ST  = (32 * 4 + 32 * 8 * 2) * 4;  // mRun,lRun,aL,mNew + pMax,pSum
constexpr int SMEM_BYTES = OFF_ST + SZ_ST;         // 295424 < 320KB

__global__ __launch_bounds__(256, 1)
void fa_fwd_kernel(const float* __restrict__ Qg, const float* __restrict__ Kg,
                   const float* __restrict__ Vg, const int* __restrict__ keyLen,
                   float* __restrict__ Og) {
  extern __shared__ char smem[];
  unsigned short* sQ  = (unsigned short*)(smem + OFF_Q);   // [BM][QSTR]
  float*          sO  = (float*)(smem + OFF_O);            // [BM][OSTR]
  unsigned short* sVt = (unsigned short*)(smem + OFF_V);   // [VQ cols][VSTR k]
  unsigned short* sP  = (unsigned short*)(smem + OFF_P);   // [BM][PSTR]
  float*          sS  = (float*)(smem + OFF_S);            // [BM][SSTR]
  float* mRun = (float*)(smem + OFF_ST);                   // [32]
  float* lRun = mRun + 32;                                 // [32]
  float* aLds = mRun + 64;                                 // [32]
  float* mNew = mRun + 96;                                 // [32]
  float* pMax = mRun + 128;                                // [32][8]
  float* pSum = mRun + 384;                                // [32][8]

  const int b     = blockIdx.y;
  const int qbase = blockIdx.x * BM;
  const int tid   = threadIdx.x;
  const int wave  = tid >> 5;
  const int lane  = tid & 31;
  const int lhalf = lane >> 4;       // 0: lanes 0-15, 1: lanes 16-31
  const int l16   = lane & 15;

  int kelen = keyLen[b];
  kelen = kelen < 0 ? 0 : (kelen > LK ? LK : kelen);
  const bool uniform = (kelen == 0);  // all-masked: softmax -> uniform (mean of V)
  const int kEnd = uniform ? LK : kelen;

  // ---- stage Q (fp32 -> bf16 packed), zero O, init stats ----
  {
    const float* Qb = Qg + ((size_t)b * LQ + qbase) * DIM;
    for (int e = tid; e < BM * (DIM / 4); e += 256) {
      int row = e >> 8;
      int c4  = (e & 255) << 2;
      float4 f = ((const float4*)(Qb + (size_t)row * DIM))[e & 255];
      uint2 w;
      w.x = pk2bf(f.x, f.y);
      w.y = pk2bf(f.z, f.w);
      *(uint2*)(sQ + row * QSTR + c4) = w;
    }
    for (int e = tid; e < BM * OSTR; e += 256) sO[e] = 0.0f;
    if (tid < 32) { mRun[tid] = -INFINITY; lRun[tid] = 0.0f; }
  }
  __syncthreads();

  const int n0 = wave << 4;          // wave's 16-key strip: 0..112

  // row-parallel softmax mapping: thread owns (row, 16 columns)
  const int srow  = tid >> 3;        // 0..31
  const int spart = tid & 7;         // 0..7  -> cols spart*16 .. +15

  for (int kv = 0; kv < kEnd; kv += BN) {
    // ---- S = Q * K^T : wave computes 32x16 strip (2 M subtiles, shared B) ----
    v8f acc0 = {0.f, 0.f, 0.f, 0.f, 0.f, 0.f, 0.f, 0.f};
    v8f acc1 = {0.f, 0.f, 0.f, 0.f, 0.f, 0.f, 0.f, 0.f};
    const int colg = kv + n0 + l16;          // this lane's key column
    if (!uniform) {
      const float* Kr = Kg + ((size_t)b * LK + colg) * DIM;
      const unsigned short* qr0 = sQ + l16 * QSTR;
      const unsigned short* qr1 = sQ + (16 + l16) * QSTR;
      #pragma unroll 2
      for (int kk = 0; kk < DIM; kk += 32) {
        BF16x16 a0U, a1U, bU;
        // A (16x32 bf16): elems 0-7 -> k=kk+lhalf*8+i, elems 8-15 -> +16
        a0U.q[0] = *(const uint4*)(qr0 + kk + lhalf * 8);
        a0U.q[1] = *(const uint4*)(qr0 + kk + 16 + lhalf * 8);
        a1U.q[0] = *(const uint4*)(qr1 + kk + lhalf * 8);
        a1U.q[1] = *(const uint4*)(qr1 + kk + 16 + lhalf * 8);
        // B (32x16 bf16): elems 0-7 -> k=kk+lhalf*8+i, elems 8-15 -> +16
        const float4* kp = (const float4*)(Kr + kk + lhalf * 8);
        float4 f0 = kp[0], f1 = kp[1], f2 = kp[4], f3 = kp[5];
        bU.w[0] = pk2bf(f0.x, f0.y); bU.w[1] = pk2bf(f0.z, f0.w);
        bU.w[2] = pk2bf(f1.x, f1.y); bU.w[3] = pk2bf(f1.z, f1.w);
        bU.w[4] = pk2bf(f2.x, f2.y); bU.w[5] = pk2bf(f2.z, f2.w);
        bU.w[6] = pk2bf(f3.x, f3.y); bU.w[7] = pk2bf(f3.z, f3.w);
        acc0 = __builtin_amdgcn_wmma_f32_16x16x32_bf16(
            false, a0U.v, false, bU.v, (short)0, acc0, false, false);
        acc1 = __builtin_amdgcn_wmma_f32_16x16x32_bf16(
            false, a1U.v, false, bU.v, (short)0, acc1, false, false);
      }
    }

    // ---- dump score strip to LDS (conflict-free) ----
    {
      float* sd = sS + n0 + l16;
      #pragma unroll
      for (int r = 0; r < 8; ++r)
        sd[(8 * lhalf + r) * SSTR] = acc0[r];
      #pragma unroll
      for (int r = 0; r < 8; ++r)
        sd[(16 + 8 * lhalf + r) * SSTR] = acc1[r];
    }
    __syncthreads();

    // ---- row-parallel softmax: thread = (row, 16 cols), scores in regs ----
    float s[16];
    {
      const float4* sr = (const float4*)(sS + srow * SSTR + spart * 16);
      float4 xa = sr[0], xb = sr[1], xc = sr[2], xd = sr[3];
      s[0] = xa.x; s[1] = xa.y; s[2]  = xa.z; s[3]  = xa.w;
      s[4] = xb.x; s[5] = xb.y; s[6]  = xb.z; s[7]  = xb.w;
      s[8] = xc.x; s[9] = xc.y; s[10] = xc.z; s[11] = xc.w;
      s[12] = xd.x; s[13] = xd.y; s[14] = xd.z; s[15] = xd.w;
    }
    const int cb = kv + spart * 16;
    #pragma unroll
    for (int j = 0; j < 16; ++j)
      if (!(uniform || cb + j < kelen)) s[j] = -INFINITY;
    {
      float mx = s[0];
      #pragma unroll
      for (int j = 1; j < 16; ++j) mx = fmaxf(mx, s[j]);
      pMax[srow * 8 + spart] = mx;
    }
    __syncthreads();

    if (tid < 32) {
      const float4* pm = (const float4*)(pMax + tid * 8);
      float4 a = pm[0], c = pm[1];
      float mb = fmaxf(fmaxf(fmaxf(a.x, a.y), fmaxf(a.z, a.w)),
                       fmaxf(fmaxf(c.x, c.y), fmaxf(c.z, c.w)));
      float mo = mRun[tid];
      float mn = fmaxf(mo, mb);
      float al = (mo == -INFINITY) ? 0.0f : __expf(mo - mn);
      mNew[tid] = mn; aLds[tid] = al;
      mRun[tid] = mn; lRun[tid] *= al;
    }
    __syncthreads();

    // ---- P = exp(S - m_new): write bf16 P tile + partial row sums ----
    {
      float mn = mNew[srow];
      float p[16];
      float sum = 0.0f;
      #pragma unroll
      for (int j = 0; j < 16; ++j) {
        p[j] = (s[j] > -INFINITY) ? __expf(s[j] - mn) : 0.0f;
        sum += p[j];
      }
      pSum[srow * 8 + spart] = sum;
      uint4 wa, wb;
      wa.x = pk2bf(p[0],  p[1]);  wa.y = pk2bf(p[2],  p[3]);
      wa.z = pk2bf(p[4],  p[5]);  wa.w = pk2bf(p[6],  p[7]);
      wb.x = pk2bf(p[8],  p[9]);  wb.y = pk2bf(p[10], p[11]);
      wb.z = pk2bf(p[12], p[13]); wb.w = pk2bf(p[14], p[15]);
      unsigned short* pd = sP + srow * PSTR + spart * 16;
      *(uint4*)(pd)     = wa;
      *(uint4*)(pd + 8) = wb;
    }
    __syncthreads();

    if (tid < 32) {
      const float4* ps = (const float4*)(pSum + tid * 8);
      float4 a = ps[0], c = ps[1];
      lRun[tid] += ((a.x + a.y) + (a.z + a.w)) + ((c.x + c.y) + (c.z + c.w));
    }

    // ---- O = O*alpha + P * V   (four 256-column quarter passes) ----
    for (int q = 0; q < 4; ++q) {
      // stage V^T quarter via 4x4 register micro-transpose (fp32 -> bf16)
      const float* Vb = Vg + ((size_t)b * LK + kv) * DIM + q * VQ;
      for (int e = tid; e < (BN / 4) * (VQ / 4); e += 256) {
        int kr4 = (e >> 6) << 2;           // k row group (0,4,..,124)
        int c4  = (e & 63) << 2;           // column group (0,4,..,252)
        const float* vb = Vb + (size_t)kr4 * DIM + c4;
        float4 r0 = *(const float4*)(vb);
        float4 r1 = *(const float4*)(vb + DIM);
        float4 r2 = *(const float4*)(vb + 2 * DIM);
        float4 r3 = *(const float4*)(vb + 3 * DIM);
        uint2 w;
        w.x = pk2bf(r0.x, r1.x); w.y = pk2bf(r2.x, r3.x);
        *(uint2*)(sVt + (c4 + 0) * VSTR + kr4) = w;
        w.x = pk2bf(r0.y, r1.y); w.y = pk2bf(r2.y, r3.y);
        *(uint2*)(sVt + (c4 + 1) * VSTR + kr4) = w;
        w.x = pk2bf(r0.z, r1.z); w.y = pk2bf(r2.z, r3.z);
        *(uint2*)(sVt + (c4 + 2) * VSTR + kr4) = w;
        w.x = pk2bf(r0.w, r1.w); w.y = pk2bf(r2.w, r3.w);
        *(uint2*)(sVt + (c4 + 3) * VSTR + kr4) = w;
      }
      __syncthreads();

      #pragma unroll
      for (int ms = 0; ms < 2; ++ms) {
        const int mbase = ms * 16;
        const unsigned short* pr = sP + (mbase + l16) * PSTR;
        BF16x16 a[4];
        #pragma unroll
        for (int j = 0; j < 4; ++j) {   // k chunks of 32 (total K = 128)
          a[j].q[0] = *(const uint4*)(pr + j * 32 + lhalf * 8);
          a[j].q[1] = *(const uint4*)(pr + j * 32 + 16 + lhalf * 8);
        }
        #pragma unroll
        for (int cs = 0; cs < 2; ++cs) {
          const int lcol = wave * 32 + cs * 16 + l16;  // col in this quarter
          const int gcol = q * VQ + lcol;              // col in O
          const unsigned short* vc = sVt + lcol * VSTR;
          BF16x16 bb[4];
          #pragma unroll
          for (int j = 0; j < 4; ++j) {
            bb[j].q[0] = *(const uint4*)(vc + j * 32 + lhalf * 8);
            bb[j].q[1] = *(const uint4*)(vc + j * 32 + 16 + lhalf * 8);
          }
          v8f c;
          #pragma unroll
          for (int r = 0; r < 8; ++r) {
            int row = mbase + 8 * lhalf + r;
            c[r] = sO[row * OSTR + gcol] * aLds[row];
          }
          #pragma unroll
          for (int j = 0; j < 4; ++j)
            c = __builtin_amdgcn_wmma_f32_16x16x32_bf16(
                false, a[j].v, false, bb[j].v, (short)0, c, false, false);
          #pragma unroll
          for (int r = 0; r < 8; ++r) {
            int row = mbase + 8 * lhalf + r;
            sO[row * OSTR + gcol] = c[r];
          }
        }
      }
      __syncthreads();
    }
  }

  // ---- normalize by l and write out ----
  __syncthreads();
  float* Ob = Og + ((size_t)b * LQ + qbase) * DIM;
  for (int e = tid; e < BM * (DIM / 4); e += 256) {
    int row = e >> 8;
    int c4  = (e & 255) << 2;
    float inv = 1.0f / lRun[row];
    const float* src = sO + row * OSTR + c4;
    float4 o;
    o.x = src[0] * inv; o.y = src[1] * inv;
    o.z = src[2] * inv; o.w = src[3] * inv;
    *(float4*)(Ob + (size_t)row * DIM + c4) = o;
  }
}

extern "C" void kernel_launch(void* const* d_in, const int* in_sizes, int n_in,
                              void* d_out, int out_size, void* d_ws, size_t ws_size,
                              hipStream_t stream) {
  (void)in_sizes; (void)n_in; (void)out_size; (void)d_ws; (void)ws_size;
  const float* Q  = (const float*)d_in[0];
  const float* K  = (const float*)d_in[1];
  const float* V  = (const float*)d_in[2];
  const int*   kl = (const int*)d_in[3];
  float* out = (float*)d_out;

  static bool attr_set = false;
  if (!attr_set) {
    (void)hipFuncSetAttribute(reinterpret_cast<const void*>(fa_fwd_kernel),
                              hipFuncAttributeMaxDynamicSharedMemorySize,
                              SMEM_BYTES);
    attr_set = true;
  }

  dim3 grid(LQ / BM, BATCH);
  fa_fwd_kernel<<<grid, 256, SMEM_BYTES, stream>>>(Q, K, V, kl, out);
}
